// Qwen3Model_34093450395983
// MI455X (gfx1250) — matrix operand order repaired and checked by reading, not compiled, not run on hardware
//
#include <hip/hip_runtime.h>

// ---------------- model dims ----------------
constexpr int Vv   = 32000;
constexpr int Dd   = 1024;
constexpr int Hh   = 16;
constexpr int HDd  = 128;
constexpr int Gg   = 8;
constexpr int HIDd = 3072;
constexpr int Ll   = 4;
constexpr int Ss   = 1024;
constexpr int Bb   = 2;
constexpr int Mm   = Bb * Ss;       // 2048 token rows

typedef __attribute__((ext_vector_type(8)))  float  v8f;
typedef __attribute__((ext_vector_type(8)))  __bf16 v8bf;
typedef __attribute__((ext_vector_type(16))) __bf16 v16bf;

__device__ __forceinline__ __bf16 f2bf(float f) {
    unsigned u = __builtin_bit_cast(unsigned, f);
    u += 0x7fffu + ((u >> 16) & 1u);                 // round-to-nearest-even
    unsigned short s = (unsigned short)(u >> 16);
    return __builtin_bit_cast(__bf16, s);
}

// Assemble a 16-element bf16 fragment from two contiguous 8-element chunks
// (A-layout and B-layout both use hi_off=16).
__device__ __forceinline__ v16bf make_frag(const __bf16* p, int hi_off) {
    v8bf lo = *(const v8bf*)p;
    v8bf hi = *(const v8bf*)(p + hi_off);
    v16bf a;
#pragma unroll
    for (int e = 0; e < 8; ++e) { a[e] = lo[e]; a[e + 8] = hi[e]; }
    return a;
}

__device__ __forceinline__ float f4comp(const float4& q, int c) {
    return (c == 0) ? q.x : (c == 1) ? q.y : (c == 2) ? q.z : q.w;
}

// =============== tiled bf16 WMMA GEMM:  C[M,N] = A_bf16[M,K] * B_f32[K,N] (+Res) ===============
// 256 threads = 8 waves; block tile 128x128, K-step 64 (2 WMMA chunks of 32);
// wave tile 32x64 (2x4 frags of 16x16); register double-buffered global->LDS staging.
__global__ __launch_bounds__(256) void gemm_bf16(
    const __bf16* __restrict__ A, const float* __restrict__ Bw,
    const float* __restrict__ Res, float* __restrict__ C,
    int Mx, int Nx, int Kx)
{
    __shared__ __bf16 Als[128][72];   // [m][k], pad 8 (row = 144B, 16B aligned)
    __shared__ __bf16 Bls[128][72];   // [n][k] col-major, pad 8

    const int tid  = threadIdx.x, lane = tid & 31, wave = tid >> 5;
    const int r    = lane & 15,   half = lane >> 4;
    const int bm   = blockIdx.y * 128, bn = blockIdx.x * 128;
    const int wm   = (wave & 3) * 32,  wn = (wave >> 2) * 64;

    // A staging: 1024 v8bf chunks -> 4 per thread
    int aR[4], aC[4];
#pragma unroll
    for (int p = 0; p < 4; ++p) {
        int c = tid + p * 256;
        aR[p] = c >> 3;          // 8 chunks per 64-wide row
        aC[p] = (c & 7) * 8;
    }
    // B staging: each thread owns 8 rows (k) x 4 cols (n)
    const int kq = tid >> 5;     // 0..7  -> rows kq*8 .. +7
    const int nq = tid & 31;     // 0..31 -> cols nq*4 .. +3

    v8f acc[2][4];
#pragma unroll
    for (int i = 0; i < 2; ++i)
#pragma unroll
        for (int j = 0; j < 4; ++j) { v8f z = {}; acc[i][j] = z; }

    v8bf  regA[4];
    float4 regB[8];
    // preload tile 0
#pragma unroll
    for (int p = 0; p < 4; ++p)
        regA[p] = *(const v8bf*)&A[(size_t)(bm + aR[p]) * Kx + aC[p]];
#pragma unroll
    for (int e = 0; e < 8; ++e)
        regB[e] = *(const float4*)&Bw[(size_t)(kq * 8 + e) * Nx + bn + nq * 4];

    for (int kt = 0; kt < Kx; kt += 64) {
        // ---- store staged registers to LDS ----
#pragma unroll
        for (int p = 0; p < 4; ++p)
            *(v8bf*)&Als[aR[p]][aC[p]] = regA[p];
#pragma unroll
        for (int c = 0; c < 4; ++c) {
            v8bf v;
#pragma unroll
            for (int e = 0; e < 8; ++e) v[e] = f2bf(f4comp(regB[e], c));
            *(v8bf*)&Bls[nq * 4 + c][kq * 8] = v;
        }
        __syncthreads();

        // ---- prefetch next tile into registers (overlaps with WMMA below) ----
        const int knext = kt + 64;
        if (knext < Kx) {
#pragma unroll
            for (int p = 0; p < 4; ++p)
                regA[p] = *(const v8bf*)&A[(size_t)(bm + aR[p]) * Kx + knext + aC[p]];
#pragma unroll
            for (int e = 0; e < 8; ++e)
                regB[e] = *(const float4*)&Bw[(size_t)(knext + kq * 8 + e) * Nx + bn + nq * 4];
        }

        // ---- compute: 2 K-chunks of 32, 8 WMMAs each ----
#pragma unroll
        for (int ck = 0; ck < 2; ++ck) {
            v16bf af[2], bfm[4];
#pragma unroll
            for (int i = 0; i < 2; ++i)
                af[i] = make_frag(&Als[wm + i * 16 + r][ck * 32 + half * 8], 16);
#pragma unroll
            for (int j = 0; j < 4; ++j)
                bfm[j] = make_frag(&Bls[wn + j * 16 + r][ck * 32 + half * 8], 16);
#pragma unroll
            for (int i = 0; i < 2; ++i)
#pragma unroll
                for (int j = 0; j < 4; ++j)
                    acc[i][j] = __builtin_amdgcn_wmma_f32_16x16x32_bf16(
                        false, af[i], false, bfm[j], (short)0, acc[i][j], false, false);
        }
        __syncthreads();
    }

    const bool hasRes = (Res != nullptr);
#pragma unroll
    for (int i = 0; i < 2; ++i)
#pragma unroll
        for (int j = 0; j < 4; ++j) {
            const int rowb = bm + wm + i * 16 + half * 8;  // C layout: VGPR e -> row rowb+e
            const int col  = bn + wn + j * 16 + r;
#pragma unroll
            for (int e = 0; e < 8; ++e) {
                size_t off = (size_t)(rowb + e) * Nx + col;
                float vv = acc[i][j][e];
                if (hasRes) vv += Res[off];
                C[off] = vv;
            }
        }
}

// =============== RMSNorm (fp32 in -> bf16 out), one block per row ===============
__global__ __launch_bounds__(256) void rmsnorm_kernel(
    const float* __restrict__ x, const float* __restrict__ scale,
    __bf16* __restrict__ out, int Dx)
{
    const int row = blockIdx.x;
    const float* xr = x + (size_t)row * Dx;
    float ss = 0.f;
    for (int i = threadIdx.x; i < Dx; i += 256) { float v = xr[i]; ss += v * v; }
#pragma unroll
    for (int off = 16; off > 0; off >>= 1) ss += __shfl_xor(ss, off, 32);
    __shared__ float red[8];
    if ((threadIdx.x & 31) == 0) red[threadIdx.x >> 5] = ss;
    __syncthreads();
    float tot = 0.f;
#pragma unroll
    for (int w = 0; w < 8; ++w) tot += red[w];
    const float rstd = rsqrtf(tot / (float)Dx + 1e-6f);
    for (int i = threadIdx.x; i < Dx; i += 256)
        out[(size_t)row * Dx + i] = f2bf(xr[i] * rstd * scale[i]);
}

// =============== per-head RMS + RoPE, transpose to [B, nh, S, HD], bf16 out ===============
__global__ __launch_bounds__(128) void rms_rope_kernel(
    const float* __restrict__ qk, const float* __restrict__ scale,
    __bf16* __restrict__ out, int nheads)
{
    const int bs = blockIdx.x / nheads;
    const int h  = blockIdx.x % nheads;
    const int b  = bs / Ss, s = bs % Ss;
    const int i  = threadIdx.x;                       // 0..127
    const float v = qk[(size_t)bs * (nheads * HDd) + h * HDd + i];
    float ss = v * v;
#pragma unroll
    for (int off = 16; off > 0; off >>= 1) ss += __shfl_xor(ss, off, 32);
    __shared__ float red[4];
    if ((i & 31) == 0) red[i >> 5] = ss;
    __syncthreads();
    const float tot = red[0] + red[1] + red[2] + red[3];
    const float xn = v * rsqrtf(tot / (float)HDd + 1e-6f) * scale[i];
    __shared__ float sh[128];
    sh[i] = xn;
    __syncthreads();
    const float partner = (i < 64) ? -sh[i + 64] : sh[i - 64];
    const int   j = (i < 64) ? i : i - 64;
    const float invfreq = __powf(1000000.0f, -(float)(2 * j) / (float)HDd);
    const float ang = (float)s * invfreq;
    const float o = xn * __cosf(ang) + partner * __sinf(ang);
    out[(((size_t)b * nheads + h) * Ss + s) * HDd + i] = f2bf(o);
}

// =============== V: fp32 [B,S,G,HD] -> bf16 [B,G,HD,S] (B-fragment friendly) ===============
__global__ __launch_bounds__(128) void vtrans_kernel(
    const float* __restrict__ v, __bf16* __restrict__ vt)
{
    const int bs = blockIdx.x / Gg;
    const int g  = blockIdx.x % Gg;
    const int b  = bs / Ss, s = bs % Ss;
    const int i  = threadIdx.x;
    vt[(((size_t)b * Gg + g) * HDd + i) * Ss + s] =
        f2bf(v[(size_t)bs * (Gg * HDd) + g * HDd + i]);
}

// =============== SiLU(g) * u -> bf16 ===============
__global__ __launch_bounds__(256) void silu_mul_kernel(
    const float* __restrict__ g, const float* __restrict__ u,
    __bf16* __restrict__ out, int n)
{
    int i = blockIdx.x * 256 + threadIdx.x;
    if (i < n) { float gv = g[i]; out[i] = f2bf((gv / (1.f + __expf(-gv))) * u[i]); }
}

// =============== embedding gather ===============
__global__ __launch_bounds__(256) void embed_kernel(
    const int* __restrict__ idx, const float* __restrict__ emb, float* __restrict__ x)
{
    const int m = blockIdx.x;
    const int t = idx[m];
    for (int i = threadIdx.x; i < Dd; i += 256)
        x[(size_t)m * Dd + i] = emb[(size_t)t * Dd + i];
}

// =============== flash attention (causal, GQA), WMMA QK^T and P*V ===============
// grid (S/128, H, B); 256 threads = 8 waves; each wave owns one 16-row Q fragment.
// K/V tile staging is register double-buffered against the WMMA/softmax work.
__global__ __launch_bounds__(256) void attn_kernel(
    const __bf16* __restrict__ Q,   // [B,H,S,HD]
    const __bf16* __restrict__ Kc,  // [B,G,S,HD]
    const __bf16* __restrict__ Vt,  // [B,G,HD,S]
    __bf16* __restrict__ Ctx)       // [B*S, H*HD]
{
    const int q0 = blockIdx.x * 128;
    const int h  = blockIdx.y;
    const int b  = blockIdx.z;
    const int g  = h / (Hh / Gg);
    const __bf16* Qb = Q  + (((size_t)b * Hh + h) * Ss) * HDd;
    const __bf16* Kb = Kc + (((size_t)b * Gg + g) * Ss) * HDd;
    const __bf16* Vb = Vt + (((size_t)b * Gg + g) * HDd) * Ss;

    __shared__ __bf16 Kls[32][HDd];      // K tile, rows=kv (B-frag col-major wrt scores)
    __shared__ __bf16 Vls[HDd][40];      // V^T tile: [hd][kv], pad 8
    __shared__ __bf16 Pls[8][16][40];    // per-wave P relayout buffer

    const int tid = threadIdx.x, lane = tid & 31, wave = tid >> 5;
    const int r = lane & 15, half = lane >> 4;
    const int qrow = q0 + wave * 16;

    // staging coordinates (2 v8bf chunks each for K and V^T per thread)
    const int c0 = tid, c1 = tid + 256;
    const int kr0 = c0 >> 4, kc0 = (c0 & 15) * 8;
    const int kr1 = c1 >> 4, kc1 = (c1 & 15) * 8;
    const int vr0 = c0 >> 2, vc0 = (c0 & 3) * 8;
    const int vr1 = c1 >> 2, vc1 = (c1 & 3) * 8;

    // preload this wave's Q fragments (A layout), 4 K-chunks over HD=128
    v16bf qf[4];
#pragma unroll
    for (int c = 0; c < 4; ++c)
        qf[c] = make_frag(Qb + (size_t)(qrow + r) * HDd + c * 32 + half * 8, 16);

    v8f acc[8];
#pragma unroll
    for (int n = 0; n < 8; ++n) { v8f z = {}; acc[n] = z; }
    float mrow[8], lrow[8];
#pragma unroll
    for (int j = 0; j < 8; ++j) { mrow[j] = -1e30f; lrow[j] = 0.f; }
    const float sc = 0.088388347648318447f;   // 1/sqrt(128)

    // preload kv tile 0 into registers
    v8bf rK0 = *(const v8bf*)&Kb[(size_t)kr0 * HDd + kc0];
    v8bf rK1 = *(const v8bf*)&Kb[(size_t)kr1 * HDd + kc1];
    v8bf rV0 = *(const v8bf*)&Vb[(size_t)vr0 * Ss + vc0];
    v8bf rV1 = *(const v8bf*)&Vb[(size_t)vr1 * Ss + vc1];

    const int kvend = q0 + 128;
    for (int kv0 = 0; kv0 < kvend; kv0 += 32) {
        // ---- store staged K/V tile to LDS ----
        *(v8bf*)&Kls[kr0][kc0] = rK0;
        *(v8bf*)&Kls[kr1][kc1] = rK1;
        *(v8bf*)&Vls[vr0][vc0] = rV0;
        *(v8bf*)&Vls[vr1][vc1] = rV1;
        __syncthreads();

        // ---- prefetch next kv tile (overlaps with math below) ----
        const int nxt = kv0 + 32;
        if (nxt < kvend) {
            rK0 = *(const v8bf*)&Kb[(size_t)(nxt + kr0) * HDd + kc0];
            rK1 = *(const v8bf*)&Kb[(size_t)(nxt + kr1) * HDd + kc1];
            rV0 = *(const v8bf*)&Vb[(size_t)vr0 * Ss + nxt + vc0];
            rV1 = *(const v8bf*)&Vb[(size_t)vr1 * Ss + nxt + vc1];
        }

        // scores: two 16x16 tiles, each 4 WMMAs over HD
        v8f sf[2];
#pragma unroll
        for (int t = 0; t < 2; ++t) {
            v8f s4 = {};
#pragma unroll
            for (int c = 0; c < 4; ++c) {
                v16bf bb = make_frag(&Kls[t * 16 + r][c * 32 + half * 8], 16);
                s4 = __builtin_amdgcn_wmma_f32_16x16x32_bf16(
                    false, qf[c], false, bb, (short)0, s4, false, false);
            }
            sf[t] = s4;
        }

        // online softmax update (row = qrow + j + half*8, col = kv0 + t*16 + r)
#pragma unroll
        for (int j = 0; j < 8; ++j) {
            const int qr = qrow + j + half * 8;
            float mx = -1e30f;
#pragma unroll
            for (int t = 0; t < 2; ++t) {
                const int col = kv0 + t * 16 + r;
                float sv = sf[t][j] * sc;
                if (col > qr) sv = -1e30f;
                sf[t][j] = sv;
                mx = fmaxf(mx, sv);
            }
#pragma unroll
            for (int off = 1; off < 16; off <<= 1) mx = fmaxf(mx, __shfl_xor(mx, off, 32));
            const float newm  = fmaxf(mrow[j], mx);
            const float alpha = __expf(mrow[j] - newm);
            float rs = 0.f;
#pragma unroll
            for (int t = 0; t < 2; ++t) {
                const float pv = __expf(sf[t][j] - newm);
                sf[t][j] = pv;
                rs += pv;
            }
#pragma unroll
            for (int off = 1; off < 16; off <<= 1) rs += __shfl_xor(rs, off, 32);
            lrow[j] = lrow[j] * alpha + rs;
            mrow[j] = newm;
#pragma unroll
            for (int n = 0; n < 8; ++n) acc[n][j] *= alpha;
        }

        // relayout P (C layout -> A layout) through LDS
#pragma unroll
        for (int t = 0; t < 2; ++t)
#pragma unroll
            for (int j = 0; j < 8; ++j)
                Pls[wave][j + half * 8][t * 16 + r] = f2bf(sf[t][j]);
        __syncthreads();

        v16bf pf = make_frag(&Pls[wave][r][half * 8], 16);
#pragma unroll
        for (int n = 0; n < 8; ++n) {
            v16bf bb = make_frag(&Vls[n * 16 + r][half * 8], 16);
            acc[n] = __builtin_amdgcn_wmma_f32_16x16x32_bf16(
                false, pf, false, bb, (short)0, acc[n], false, false);
        }
        __syncthreads();
    }

    // normalize and write ctx (bf16, [B*S, H*HD])
#pragma unroll
    for (int n = 0; n < 8; ++n)
#pragma unroll
        for (int j = 0; j < 8; ++j) {
            const int qr = qrow + j + half * 8;
            const float o = acc[n][j] / lrow[j];
            Ctx[((size_t)b * Ss + qr) * (Hh * HDd) + h * HDd + n * 16 + r] = f2bf(o);
        }
}

// =========================== host orchestration ===========================
extern "C" void kernel_launch(void* const* d_in, const int* in_sizes, int n_in,
                              void* d_out, int out_size, void* d_ws, size_t ws_size,
                              hipStream_t stream)
{
    (void)in_sizes; (void)n_in; (void)out_size; (void)ws_size;
    const int*   in_idx    = (const int*)  d_in[0];
    const float* tok_emb   = (const float*)d_in[1];
    const float* Wq        = (const float*)d_in[2];
    const float* Wk        = (const float*)d_in[3];
    const float* Wv        = (const float*)d_in[4];
    const float* Wo        = (const float*)d_in[5];
    const float* q_scale   = (const float*)d_in[6];
    const float* k_scale   = (const float*)d_in[7];
    const float* n1_scale  = (const float*)d_in[8];
    const float* n2_scale  = (const float*)d_in[9];
    const float* fc1       = (const float*)d_in[10];
    const float* fc2       = (const float*)d_in[11];
    const float* fc3       = (const float*)d_in[12];
    const float* fin_scale = (const float*)d_in[13];
    const float* out_head  = (const float*)d_in[14];
    float* logits = (float*)d_out;

    char* wp = (char*)d_ws;
    auto take = [&](size_t bytes) -> void* {
        void* p = (void*)wp;
        wp += (bytes + 255) & ~(size_t)255;
        return p;
    };
    float*  x     = (float*) take((size_t)Mm * Dd * 4);
    __bf16* hbf   = (__bf16*)take((size_t)Mm * Dd * 2);
    float*  qbuf  = (float*) take((size_t)Mm * Hh * HDd * 4);
    float*  kbuf  = (float*) take((size_t)Mm * Gg * HDd * 4);
    float*  vbuf  = (float*) take((size_t)Mm * Gg * HDd * 4);
    __bf16* qbf   = (__bf16*)take((size_t)Mm * Hh * HDd * 2);
    __bf16* kbf   = (__bf16*)take((size_t)Mm * Gg * HDd * 2);
    __bf16* vtbf  = (__bf16*)take((size_t)Mm * Gg * HDd * 2);
    __bf16* ctxbf = (__bf16*)take((size_t)Mm * Hh * HDd * 2);
    float*  gbuf  = (float*) take((size_t)Mm * HIDd * 4);
    float*  ubuf  = (float*) take((size_t)Mm * HIDd * 4);
    __bf16* actbf = (__bf16*)take((size_t)Mm * HIDd * 2);

    embed_kernel<<<Mm, 256, 0, stream>>>(in_idx, tok_emb, x);

    for (int l = 0; l < Ll; ++l) {
        const float* wq = Wq  + (size_t)l * Dd * (Hh * HDd);
        const float* wk = Wk  + (size_t)l * Dd * (Gg * HDd);
        const float* wv = Wv  + (size_t)l * Dd * (Gg * HDd);
        const float* wo = Wo  + (size_t)l * (Hh * HDd) * Dd;
        const float* f1 = fc1 + (size_t)l * Dd * HIDd;
        const float* f2 = fc2 + (size_t)l * Dd * HIDd;
        const float* f3 = fc3 + (size_t)l * HIDd * Dd;

        rmsnorm_kernel<<<Mm, 256, 0, stream>>>(x, n1_scale + (size_t)l * Dd, hbf, Dd);
        gemm_bf16<<<dim3(Hh * HDd / 128, Mm / 128), 256, 0, stream>>>(hbf, wq, nullptr, qbuf, Mm, Hh * HDd, Dd);
        gemm_bf16<<<dim3(Gg * HDd / 128, Mm / 128), 256, 0, stream>>>(hbf, wk, nullptr, kbuf, Mm, Gg * HDd, Dd);
        gemm_bf16<<<dim3(Gg * HDd / 128, Mm / 128), 256, 0, stream>>>(hbf, wv, nullptr, vbuf, Mm, Gg * HDd, Dd);
        rms_rope_kernel<<<Mm * Hh, 128, 0, stream>>>(qbuf, q_scale + (size_t)l * HDd, qbf, Hh);
        rms_rope_kernel<<<Mm * Gg, 128, 0, stream>>>(kbuf, k_scale + (size_t)l * HDd, kbf, Gg);
        vtrans_kernel<<<Mm * Gg, 128, 0, stream>>>(vbuf, vtbf);
        attn_kernel<<<dim3(Ss / 128, Hh, Bb), 256, 0, stream>>>(qbf, kbf, vtbf, ctxbf);
        gemm_bf16<<<dim3(Dd / 128, Mm / 128), 256, 0, stream>>>(ctxbf, wo, x, x, Mm, Dd, Hh * HDd);
        rmsnorm_kernel<<<Mm, 256, 0, stream>>>(x, n2_scale + (size_t)l * Dd, hbf, Dd);
        gemm_bf16<<<dim3(HIDd / 128, Mm / 128), 256, 0, stream>>>(hbf, f1, nullptr, gbuf, Mm, HIDd, Dd);
        gemm_bf16<<<dim3(HIDd / 128, Mm / 128), 256, 0, stream>>>(hbf, f2, nullptr, ubuf, Mm, HIDd, Dd);
        silu_mul_kernel<<<(Mm * HIDd + 255) / 256, 256, 0, stream>>>(gbuf, ubuf, actbf, Mm * HIDd);
        gemm_bf16<<<dim3(Dd / 128, Mm / 128), 256, 0, stream>>>(actbf, f3, x, x, Mm, Dd, HIDd);
    }

    rmsnorm_kernel<<<Mm, 256, 0, stream>>>(x, fin_scale, hbf, Dd);
    gemm_bf16<<<dim3(Vv / 128, Mm / 128), 256, 0, stream>>>(hbf, out_head, nullptr, logits, Mm, Vv, Dd);
}
